// MarkerRenderer_51316269252922
// MI455X (gfx1250) — matrix operand order repaired and checked by reading, not compiled, not run on hardware
//
#include <hip/hip_runtime.h>
#include <hip/hip_bf16.h>

typedef __attribute__((ext_vector_type(16))) _Float16 v16h;
typedef __attribute__((ext_vector_type(8)))  float    v8f;

#define HH 736
#define WW 1280
constexpr int HW   = HH * WW;
constexpr int NM   = 8;      // markers
constexpr int NBRD = 4;      // boards
constexpr int MFW  = 260;    // marker filtered size (256+8-4)
constexpr int BFW  = 516;    // board  filtered size (512+8-4)
constexpr int MPAD = 264;    // padded upsampled marker
constexpr int MPLANE = MFW * MFW;       // 67600
constexpr int BPLANE = BFW * BFW;       // 266256

// workspace layout (float units)
constexpr int OFF_HINV = 0;                       // 12*9 floats
constexpr int OFF_BB   = 128;                     // 12*4 ints (aliased)
constexpr int OFF_SP   = 176;                     // 4*3 floats
constexpr int OFF_KW   = 192;                     // 25 floats
constexpr int OFF_FAM  = 256;                     // 260*260
constexpr int OFF_FAB  = OFF_FAM + MPLANE;        // 516*516
constexpr int OFF_FPM  = OFF_FAB + BPLANE;        // 8*3*260*260
constexpr int OFF_PAD  = OFF_FPM + NM * 3 * MPLANE; // f16 region: 24*264*264 halves

__device__ inline float render_spec(float lx, float ly, float lz,
                                    float vx, float vy, float vz,
                                    float nx, float ny, float nz, float rough) {
  const float PI = 3.14159265358979f;
  float mx = (lx + vx) * 0.5f, my = (ly + vy) * 0.5f, mz = (lz + vz) * 0.5f;
  float len = fmaxf(sqrtf(mx * mx + my * my + mz * mz), 1e-8f);
  float hx = mx / len, hy = my / len, hz = mz / len;
  float NoH = fmaxf(nx * hx + ny * hy + nz * hz, 0.f);
  float NoV = fmaxf(nx * vx + ny * vy + nz * vz, 0.f);
  float NoL = fmaxf(nx * lx + ny * ly + nz * lz, 0.f);
  float VoH = fmaxf(vx * hx + vy * hy + vz * hz, 0.f);
  float a = rough * rough;
  float tmp = a / fmaxf(NoH * NoH * (a * a - 1.f) + 1.f, 1e-8f);
  float D = tmp * tmp * (1.f / PI);
  float kk = a * 0.5f;
  float G = NoL * NoV / fmaxf((NoL * (1.f - kk) + kk) * (NoV * (1.f - kk) + kk), 1e-8f);
  float coeff = VoH * (-5.55473f * VoH - 6.98316f);
  float F = 0.1f + 0.9f * exp2f(coeff);
  float fs = D * G * F / fmaxf(4.f * NoL * NoV, 1e-8f);
  return fs * NoL * PI;
}

// ---- setup: disk kernel, homography inverses (scaled), bboxes, spec power
__global__ void setup_kernel(const float* __restrict__ mh, const float* __restrict__ bh,
                             const float* __restrict__ refl, const float* __restrict__ bvd,
                             const float* __restrict__ bn, const float* __restrict__ avg,
                             const float* __restrict__ bmax, const float* __restrict__ rough,
                             const float* __restrict__ sscale, float* __restrict__ ws) {
  int t = threadIdx.x;
  if (t == 0) {
    float tmpw[25]; float s = 0.f;
    for (int dy = -2; dy <= 2; dy++)
      for (int dx = -2; dx <= 2; dx++) {
        float d = sqrtf((float)(dx * dx + dy * dy));
        float v = 2.5f - d; v = v < 0.f ? 0.f : (v > 1.f ? 1.f : v);
        tmpw[(dy + 2) * 5 + dx + 2] = v; s += v;
      }
    for (int i = 0; i < 25; i++) ws[OFF_KW + i] = tmpw[i] / s;
    for (int i = 25; i < 32; i++) ws[OFF_KW + i] = 0.f;
  }
  if (t < 12) {
    const float* h = (t < NM) ? (mh + t * 9) : (bh + (t - NM) * 9);
    float det = h[0] * (h[4] * h[8] - h[5] * h[7])
              - h[1] * (h[3] * h[8] - h[5] * h[6])
              + h[2] * (h[3] * h[7] - h[4] * h[6]);
    float inv[9];
    inv[0] = (h[4] * h[8] - h[5] * h[7]) / det;
    inv[1] = (h[2] * h[7] - h[1] * h[8]) / det;
    inv[2] = (h[1] * h[5] - h[2] * h[4]) / det;
    inv[3] = (h[5] * h[6] - h[3] * h[8]) / det;
    inv[4] = (h[0] * h[8] - h[2] * h[6]) / det;
    inv[5] = (h[2] * h[3] - h[0] * h[5]) / det;
    inv[6] = (h[3] * h[7] - h[4] * h[6]) / det;
    inv[7] = (h[1] * h[6] - h[0] * h[7]) / det;
    inv[8] = (h[0] * h[4] - h[1] * h[3]) / det;
    float sca = (t < NM) ? (256.f / 260.f) : (512.f / 516.f);
    for (int i = 0; i < 6; i++) ws[OFF_HINV + t * 9 + i] = inv[i] * sca;
    for (int i = 6; i < 9; i++) ws[OFF_HINV + t * 9 + i] = inv[i];
    // bbox from forward transform of corners
    const float cxs[4] = {-1.f, 1.f, 1.f, -1.f};
    const float cys[4] = {-1.f, -1.f, 1.f, 1.f};
    float mnx = 1e30f, mny = 1e30f, mxx = -1e30f, mxy = -1e30f;
    for (int k = 0; k < 4; k++) {
      float cx = cxs[k], cy = cys[k];
      float tz = h[6] * cx + h[7] * cy + h[8];
      float sc = fabsf(tz) > 1e-8f ? 1.f / (tz + 1e-8f) : 1.f;
      float X = sc * (h[0] * cx + h[1] * cy + h[2]);
      float Y = sc * (h[3] * cx + h[4] * cy + h[5]);
      mnx = fminf(mnx, X); mny = fminf(mny, Y);
      mxx = fmaxf(mxx, X); mxy = fmaxf(mxy, Y);
    }
    float lox = fminf(fmaxf(mnx - 2.f, 0.f), (float)WW);
    float loy = fminf(fmaxf(mny - 2.f, 0.f), (float)HH);
    float hix = fminf(fmaxf(mxx + 2.f, 0.f), (float)WW);
    float hiy = fminf(fmaxf(mxy + 2.f, 0.f), (float)HH);
    int* bb = (int*)ws;
    bb[OFF_BB + t * 4 + 0] = (int)floorf(lox);
    bb[OFF_BB + t * 4 + 1] = (int)floorf(loy);
    bb[OFF_BB + t * 4 + 2] = (int)floorf(hix);
    bb[OFF_BB + t * 4 + 3] = (int)floorf(hiy);
  }
  if (t >= 12 && t < 16) {
    int n = t - 12;
    float sm = render_spec(refl[n * 3], refl[n * 3 + 1], refl[n * 3 + 2],
                           bvd[n * 3], bvd[n * 3 + 1], bvd[n * 3 + 2],
                           bn[n * 3], bn[n * 3 + 1], bn[n * 3 + 2], rough[n]);
    float f = (1.1f - bmax[n]) * sscale[n] / (sm + 1e-8f);
    for (int c = 0; c < 3; c++) ws[OFF_SP + n * 3 + c] = avg[n * 3 + c] * f;
  }
}

// ---- nearest-upsample x4 + edge-pad(4) markers into f16 (264x264 per n,c)
__global__ void pad_markers_kernel(const float* __restrict__ markers, _Float16* __restrict__ pad) {
  int idx = blockIdx.x * blockDim.x + threadIdx.x;
  if (idx >= 24 * MPAD * MPAD) return;
  int nc = idx / (MPAD * MPAD);
  int rem = idx - nc * (MPAD * MPAD);
  int r = rem / MPAD, col = rem - r * MPAD;
  int uy = r - 4;  uy = uy < 0 ? 0 : (uy > 255 ? 255 : uy);
  int ux = col - 4; ux = ux < 0 ? 0 : (ux > 255 ? 255 : ux);
  float v = markers[(nc * 64 + (uy >> 2)) * 64 + (ux >> 2)];
  pad[idx] = (_Float16)v;
}

// ---- alpha planes (blur of ones with zero pad) for marker(260) and board(516)
__global__ void fa_kernel(const float* __restrict__ kw, float* __restrict__ fam, float* __restrict__ fab) {
  int idx = blockIdx.x * blockDim.x + threadIdx.x;
  if (idx >= MPLANE + BPLANE) return;
  int lim, x, y; float* dst; int j;
  if (idx < MPLANE) { lim = MFW; j = idx; y = j / MFW; x = j - y * MFW; dst = fam; }
  else { lim = BFW; j = idx - MPLANE; y = j / BFW; x = j - y * BFW; dst = fab; }
  float s = 0.f;
  for (int dy = 0; dy < 5; dy++) {
    int ry = y + dy;
    bool oy = (ry >= 4) && (ry < lim);
    for (int dx = 0; dx < 5; dx++) {
      int rx = x + dx;
      if (oy && rx >= 4 && rx < lim) s += kw[dy * 5 + dx];
    }
  }
  dst[j] = s;
}

// ---- 5x5 blur as im2col WMMA: one wave -> 16 output pixels of one row.
// A(16x32 f16): rows = output x positions, K = 25 taps (padded to 32, zeros in B).
// A element mapping (16-bit A 16x32): lane<16: e<8 -> K=e, else K=e+8;
//                                     lane>=16: +8.
// B element mapping (16-bit B 32x16, same as A): e<8 -> K=e, else K=e+8; +8*hi.
// Weights broadcast across columns -> every column of D holds the result;
// lanes 0 and 16 write rows 0..7 / 8..15 as two float4 stores.
// Last x-tile overlaps (x0 clamped to 244) so no store guard is needed.
__global__ __launch_bounds__(32) void blur_wmma_kernel(const _Float16* __restrict__ pad,
                                                       const float* __restrict__ kw,
                                                       float* __restrict__ fp) {
  int lane = threadIdx.x;
  int x0 = blockIdx.x * 16;
  if (x0 > MFW - 16) x0 = MFW - 16;  // 244: overlap, identical recompute
  int y  = blockIdx.y;               // 0..259
  int nc = blockIdx.z;               // 0..23
  const _Float16* src = pad + nc * (MPAD * MPAD) + y * MPAD;
  int M  = lane & 15;
  int hi = lane >> 4;
  v16h a, b;
#pragma unroll
  for (int e = 0; e < 16; e++) {
    int Ka = (e < 8 ? e : e + 8) + hi * 8;
    int Kc = Ka < 25 ? Ka : 24;        // clamp: product is zeroed via B anyway
    int ky = Kc / 5, kx = Kc - ky * 5;
    a[e] = src[ky * MPAD + x0 + M + kx];
    int Kb = (e < 8 ? e : e + 8) + hi * 8;
    float wv = kw[Kb];                  // kw[25..31] initialized to 0 in setup
    b[e] = (_Float16)wv;
  }
  v8f c = {};
  c = __builtin_amdgcn_wmma_f32_16x16x32_f16(false, a, false, b, (short)0, c, false, false);
  if ((lane & 15) == 0) {
    float* dst = fp + nc * MPLANE + y * MFW + x0 + hi * 8;
    float4 lo4 = make_float4(c[0], c[1], c[2], c[3]);
    float4 hi4 = make_float4(c[4], c[5], c[6], c[7]);
    *(float4*)dst = lo4;
    *(float4*)(dst + 4) = hi4;
  }
}

// ---- fused per-pixel composite + specular
__global__ __launch_bounds__(256) void composite_kernel(
    const float* __restrict__ image, const float* __restrict__ vd,
    const float* __restrict__ noise, const float* __restrict__ ws,
    const float* __restrict__ refl, const float* __restrict__ bnrm,
    const float* __restrict__ rough, const float* __restrict__ diff,
    float* __restrict__ out) {
  __shared__ float s_hinv[108];
  __shared__ int   s_bb[48];
  __shared__ float s_sp[12], s_refl[12], s_bn[12], s_diff[12], s_rough[4];
  int tid = threadIdx.x;
  if (tid < 108) s_hinv[tid] = ws[OFF_HINV + tid];
  if (tid < 48)  s_bb[tid]   = ((const int*)ws)[OFF_BB + tid];
  if (tid < 12) {
    s_sp[tid] = ws[OFF_SP + tid];
    s_refl[tid] = refl[tid]; s_bn[tid] = bnrm[tid]; s_diff[tid] = diff[tid];
  }
  if (tid < 4) s_rough[tid] = rough[tid];
  __syncthreads();

  int idx = blockIdx.x * 256 + tid;
  if (idx >= HW) return;
  int y = idx / WW, x = idx - y * WW;
  float px = x + 0.5f, py = y + 0.5f;

  const float* fam = ws + OFF_FAM;
  const float* fab = ws + OFF_FAB;
  const float* fpm = ws + OFF_FPM;

  // markers
  float pm0 = 0.f, pm1 = 0.f, pm2 = 0.f, pma = 0.f;
  for (int n = 0; n < NM; n++) {
    const int* bb = &s_bb[n * 4];
    if (x < bb[0] || x >= bb[2] || y < bb[1] || y >= bb[3]) continue;
    const float* h = &s_hinv[n * 9];
    float tz = h[6] * px + h[7] * py + h[8];
    float sc = fabsf(tz) > 1e-8f ? 1.f / (tz + 1e-8f) : 1.f;
    float gx = (sc * (h[0] * px + h[1] * py + h[2]) + 1.f) * (0.5f * MFW) - 0.5f;
    float gy = (sc * (h[3] * px + h[4] * py + h[5]) + 1.f) * (0.5f * MFW) - 0.5f;
    float fx0 = floorf(gx), fy0 = floorf(gy);
    float wx = gx - fx0, wy = gy - fy0;
    const float* p0 = fpm + n * 3 * MPLANE;
#pragma unroll
    for (int t = 0; t < 4; t++) {
      float xf = fx0 + (float)(t & 1), yf = fy0 + (float)(t >> 1);
      if (xf < 0.f || xf >= (float)MFW || yf < 0.f || yf >= (float)MFW) continue;
      float wt = ((t & 1) ? wx : (1.f - wx)) * ((t >> 1) ? wy : (1.f - wy));
      int b = (int)yf * MFW + (int)xf;
      pm0 += p0[b] * wt;
      pm1 += p0[MPLANE + b] * wt;
      pm2 += p0[2 * MPLANE + b] * wt;
      pma += fam[b] * wt;
    }
  }

  // boards (color plane is analytically diffuse * in-bounds weight)
  float pb0 = 0.f, pb1 = 0.f, pb2 = 0.f, pba = 0.f;
  float sd0 = 0.f, sd1 = 0.f, sd2 = 0.f;
  float sn0 = 0.f, sn1 = 0.f, sn2 = 0.f, rg = 0.f;
  float sp0 = 0.f, sp1 = 0.f, sp2 = 0.f;
  for (int nb = 0; nb < NBRD; nb++) {
    int pp = NM + nb;
    const int* bb = &s_bb[pp * 4];
    if (x < bb[0] || x >= bb[2] || y < bb[1] || y >= bb[3]) continue;
    const float* h = &s_hinv[pp * 9];
    float tz = h[6] * px + h[7] * py + h[8];
    float sc = fabsf(tz) > 1e-8f ? 1.f / (tz + 1e-8f) : 1.f;
    float gx = (sc * (h[0] * px + h[1] * py + h[2]) + 1.f) * (0.5f * BFW) - 0.5f;
    float gy = (sc * (h[3] * px + h[4] * py + h[5]) + 1.f) * (0.5f * BFW) - 0.5f;
    float fx0 = floorf(gx), fy0 = floorf(gy);
    float wx = gx - fx0, wy = gy - fy0;
    float okw = 0.f, aa = 0.f;
#pragma unroll
    for (int t = 0; t < 4; t++) {
      float xf = fx0 + (float)(t & 1), yf = fy0 + (float)(t >> 1);
      if (xf < 0.f || xf >= (float)BFW || yf < 0.f || yf >= (float)BFW) continue;
      float wt = ((t & 1) ? wx : (1.f - wx)) * ((t >> 1) ? wy : (1.f - wy));
      int b = (int)yf * BFW + (int)xf;
      okw += wt;
      aa  += fab[b] * wt;
    }
    pb0 += s_diff[nb * 3 + 0] * okw;
    pb1 += s_diff[nb * 3 + 1] * okw;
    pb2 += s_diff[nb * 3 + 2] * okw;
    pba += aa;
    sp0 += aa * s_sp[nb * 3 + 0];
    sp1 += aa * s_sp[nb * 3 + 1];
    sp2 += aa * s_sp[nb * 3 + 2];
    if (aa != 0.f) {
      sd0 += s_refl[nb * 3 + 0]; sd1 += s_refl[nb * 3 + 1]; sd2 += s_refl[nb * 3 + 2];
      sn0 += s_bn[nb * 3 + 0];   sn1 += s_bn[nb * 3 + 1];   sn2 += s_bn[nb * 3 + 2];
      rg  += s_rough[nb];
    }
  }

  float n0 = sn0 + noise[idx];
  float n1 = sn1 + noise[HW + idx];
  float n2 = sn2 + noise[2 * HW + idx];
  float ln = fmaxf(sqrtf(n0 * n0 + n1 * n1 + n2 * n2), 1e-8f);
  n0 /= ln; n1 /= ln; n2 /= ln;
  float v0 = vd[idx * 3 + 0], v1 = vd[idx * 3 + 1], v2 = vd[idx * 3 + 2];
  float spec = render_spec(sd0, sd1, sd2, v0, v1, v2, n0, n1, n2, rg);

  float i0 = image[idx], i1 = image[HW + idx], i2 = image[2 * HW + idx];
  float st0 = pb0 * (1.f - pma) + pb0 * pm0 * pma;
  float st1 = pb1 * (1.f - pma) + pb1 * pm1 * pma;
  float st2 = pb2 * (1.f - pma) + pb2 * pm2 * pma;
  out[idx]          = i0 * (1.f - pba) + i0 * pba * st0 + spec * sp0;
  out[HW + idx]     = i1 * (1.f - pba) + i1 * pba * st1 + spec * sp1;
  out[2 * HW + idx] = i2 * (1.f - pba) + i2 * pba * st2 + spec * sp2;
}

extern "C" void kernel_launch(void* const* d_in, const int* in_sizes, int n_in,
                              void* d_out, int out_size, void* d_ws, size_t ws_size,
                              hipStream_t stream) {
  const float* image   = (const float*)d_in[0];
  const float* vdirs   = (const float*)d_in[1];
  const float* markers = (const float*)d_in[2];
  const float* mh      = (const float*)d_in[3];
  const float* bh      = (const float*)d_in[4];
  const float* refl    = (const float*)d_in[5];
  const float* bvd     = (const float*)d_in[6];
  const float* bn      = (const float*)d_in[7];
  const float* avg     = (const float*)d_in[8];
  const float* bmax    = (const float*)d_in[9];
  const float* rough   = (const float*)d_in[10];
  const float* diff    = (const float*)d_in[11];
  const float* sscale  = (const float*)d_in[12];
  const float* noise   = (const float*)d_in[13];
  float* ws  = (float*)d_ws;
  float* out = (float*)d_out;

  setup_kernel<<<1, 32, 0, stream>>>(mh, bh, refl, bvd, bn, avg, bmax, rough, sscale, ws);

  int padN = 24 * MPAD * MPAD;
  pad_markers_kernel<<<(padN + 255) / 256, 256, 0, stream>>>(markers, (_Float16*)(ws + OFF_PAD));

  int faN = MPLANE + BPLANE;
  fa_kernel<<<(faN + 255) / 256, 256, 0, stream>>>(ws + OFF_KW, ws + OFF_FAM, ws + OFF_FAB);

  blur_wmma_kernel<<<dim3(17, MFW, 24), 32, 0, stream>>>(
      (const _Float16*)(ws + OFF_PAD), ws + OFF_KW, ws + OFF_FPM);

  composite_kernel<<<(HW + 255) / 256, 256, 0, stream>>>(
      image, vdirs, noise, ws, refl, bn, rough, diff, out);
}
